// UnifiedMemoryLayer_32091995636208
// MI455X (gfx1250) — matrix operand order repaired and checked by reading, not compiled, not run on hardware
//
#include <hip/hip_runtime.h>
#include <hip/hip_bf16.h>

// ---------------------------------------------------------------------------
// CDNA5 (gfx1250) wave32 bf16-WMMA implementation of UnifiedMemoryLayer.
// 128x128 macro-tiles, 8 waves, 8 x v_wmma_f32_16x16x32_bf16 per wave per
// K-step, double-buffered LDS staging, bf16 pre-packed weights.
// ---------------------------------------------------------------------------

typedef __bf16 bf16_t;
typedef bf16_t v16bf __attribute__((ext_vector_type(16)));
typedef float  v8f   __attribute__((ext_vector_type(8)));

#define TILE_M 128
#define TILE_N 128
#define TILE_K 32

__device__ __forceinline__ unsigned short f32_bf16(float f) {
  union { float f; unsigned int u; } c; c.f = f;
  unsigned int u = c.u;
  u += 0x7FFFu + ((u >> 16) & 1u);   // round-to-nearest-even
  return (unsigned short)(u >> 16);
}

// Generic batched GEMM:  C[z] = act(alpha * A[z] @ op(B[z]) + bias + R[z])
// z = blockIdx.z decomposed as (batch = z/nH, head = z%nH).
// A is f32; B is f32 or pre-packed bf16 (bIsBf16).
__global__ void __launch_bounds__(256)
gemm_bf16_wmma(const float* __restrict__ A, int lda, long long sA, long long hA,
               const void* __restrict__ Bm, int ldb, long long sB, long long hB,
               int transB, int bIsBf16,
               const float* __restrict__ bias,
               const float* __restrict__ Rsd, int ldr, long long sR, long long hR,
               float* __restrict__ C, int ldc, long long sC, long long hC,
               int M, int N, int K, float alpha, int act, int nH)
{
  __shared__ unsigned short sa[2][TILE_M * TILE_K];  // [m][k] bf16
  __shared__ unsigned short sb[2][TILE_N * TILE_K];  // [n][k] bf16 (B transposed in LDS)

  const int z  = blockIdx.z;
  const int bz = z / nH;
  const int hz = z % nH;
  const long long bOff = (long long)bz * sB + (long long)hz * hB;
  const float*          Ab = A + (long long)bz * sA + (long long)hz * hA;
  const float*          Bf = (const float*)Bm + bOff;
  const unsigned short* Bh = (const unsigned short*)Bm + bOff;
  const float* Rb = Rsd ? (Rsd + (long long)bz * sR + (long long)hz * hR) : nullptr;
  float*       Cb = C + (long long)bz * sC + (long long)hz * hC;

  const int row0 = blockIdx.y * TILE_M;
  const int col0 = blockIdx.x * TILE_N;
  const int tid  = threadIdx.x;
  const int lane = tid & 31;
  const int wave = tid >> 5;          // 8 waves: 4 (rows) x 2 (cols)
  const int wm   = (wave >> 1) * 32;  // wave sub-tile: 32 rows x 64 cols
  const int wn   = (wave & 1) * 64;
  const int g    = lane >> 4;
  const int l16  = lane & 15;
  const bool nEdge = (col0 + TILE_N > N);

  // ---- staging: global -> LDS (f32->bf16 for A / f32-B; raw copy for bf16-B)
  auto stage = [&](int bi, int k0) {
    unsigned int* sau = (unsigned int*)sa[bi];
    unsigned int* sbu = (unsigned int*)sb[bi];
    // A tile [128 x 32] via float4 (M always a multiple of TILE_M here)
#pragma unroll
    for (int i = 0; i < 4; ++i) {
      int f = tid + i * 256;              // 1024 float4s
      int r = f >> 3, k4 = (f & 7) * 4;
      float4 v4 = *(const float4*)(Ab + (long long)(row0 + r) * lda + (k0 + k4));
      sau[r * 16 + (k4 >> 1)]     = ((unsigned int)f32_bf16(v4.y) << 16) | f32_bf16(v4.x);
      sau[r * 16 + (k4 >> 1) + 1] = ((unsigned int)f32_bf16(v4.w) << 16) | f32_bf16(v4.z);
    }
    // B tile -> sb[n][k]
    if (transB) {                         // B is [N,K]; contiguous along k; clamp row
      if (bIsBf16) {
#pragma unroll
        for (int i = 0; i < 4; ++i) {
          int f = tid + i * 256;
          int n = f >> 3, k4 = (f & 7) * 4;
          int gc = col0 + n; if (gc >= N) gc = N - 1;
          const unsigned short* src = Bh + (long long)gc * ldb + (k0 + k4);
          sbu[n * 16 + (k4 >> 1)]     = *(const unsigned int*)(src);
          sbu[n * 16 + (k4 >> 1) + 1] = *(const unsigned int*)(src + 2);
        }
      } else {
#pragma unroll
        for (int i = 0; i < 4; ++i) {
          int f = tid + i * 256;
          int n = f >> 3, k4 = (f & 7) * 4;
          int gc = col0 + n; if (gc >= N) gc = N - 1;
          float4 v4 = *(const float4*)(Bf + (long long)gc * ldb + (k0 + k4));
          sbu[n * 16 + (k4 >> 1)]     = ((unsigned int)f32_bf16(v4.y) << 16) | f32_bf16(v4.x);
          sbu[n * 16 + (k4 >> 1) + 1] = ((unsigned int)f32_bf16(v4.w) << 16) | f32_bf16(v4.z);
        }
      }
    } else {                              // B is [K,N]; contiguous along n
      if (!nEdge) {
        if (bIsBf16) {
#pragma unroll
          for (int i = 0; i < 4; ++i) {
            int f = tid + i * 256;        // 32 k x 32 n-quads
            int k = f >> 5, n4 = (f & 31) * 4;
            uint2 q = *(const uint2*)(Bh + (long long)(k0 + k) * ldb + col0 + n4);
            sb[bi][(n4 + 0) * 32 + k] = (unsigned short)(q.x & 0xffffu);
            sb[bi][(n4 + 1) * 32 + k] = (unsigned short)(q.x >> 16);
            sb[bi][(n4 + 2) * 32 + k] = (unsigned short)(q.y & 0xffffu);
            sb[bi][(n4 + 3) * 32 + k] = (unsigned short)(q.y >> 16);
          }
        } else {
#pragma unroll
          for (int i = 0; i < 4; ++i) {
            int f = tid + i * 256;
            int k = f >> 5, n4 = (f & 31) * 4;
            float4 v4 = *(const float4*)(Bf + (long long)(k0 + k) * ldb + col0 + n4);
            sb[bi][(n4 + 0) * 32 + k] = f32_bf16(v4.x);
            sb[bi][(n4 + 1) * 32 + k] = f32_bf16(v4.y);
            sb[bi][(n4 + 2) * 32 + k] = f32_bf16(v4.z);
            sb[bi][(n4 + 3) * 32 + k] = f32_bf16(v4.w);
          }
        }
      } else {                            // column-edge fallback (N=320/64 tiles)
#pragma unroll
        for (int i = 0; i < 16; ++i) {
          int idx = tid + i * 256;        // 4096 elements
          int n = idx >> 5, kk = idx & 31;
          int gc = col0 + n; if (gc >= N) gc = N - 1;
          long long src = (long long)(k0 + kk) * ldb + gc;
          sb[bi][n * 32 + kk] = bIsBf16 ? Bh[src] : f32_bf16(Bf[src]);
        }
      }
    }
  };

  v8f acc[2][4];
#pragma unroll
  for (int i = 0; i < 2; ++i)
#pragma unroll
    for (int j = 0; j < 4; ++j)
#pragma unroll
      for (int r = 0; r < 8; ++r) acc[i][j][r] = 0.0f;

  stage(0, 0);
  int buf = 0;
  for (int k0 = 0; k0 < K; k0 += TILE_K) {
    __syncthreads();

    // 1) fragment loads for this step (ds_loads; documented wave32 layouts)
    union Frag { v16bf v; unsigned int u[8]; } af[2], bfb[4];
#pragma unroll
    for (int mi = 0; mi < 2; ++mi) {
      const unsigned short* base = &sa[buf][(wm + mi * 16 + l16) * TILE_K];
      // A 16x32 bf16: dword p holds K = 8g + 2(p&3) + 16(p>>2), K+1
#pragma unroll
      for (int p = 0; p < 8; ++p)
        af[mi].u[p] = *(const unsigned int*)(base + 8 * g + 2 * (p & 3) + ((p >> 2) * 16));
    }
#pragma unroll
    for (int ni = 0; ni < 4; ++ni) {
      const unsigned short* base = &sb[buf][(wn + ni * 16 + l16) * TILE_K];
      // B 32x16 bf16: dword p holds K = 8g + 2(p&3) + 16(p>>2), K+1 (N = lane&15)
#pragma unroll
      for (int p = 0; p < 8; ++p)
        bfb[ni].u[p] = *(const unsigned int*)(base + 8 * g + 2 * (p & 3) + ((p >> 2) * 16));
    }

    // 2) stage next tile (global loads + ds_stores overlap the WMMAs below)
    if (k0 + TILE_K < K) stage(buf ^ 1, k0 + TILE_K);
    if (k0 + 2 * TILE_K < K) {  // prefetch tile after next (global_prefetch_b8)
      __builtin_prefetch(Ab + (long long)(row0 + (tid & 127)) * lda + k0 + 2 * TILE_K, 0, 0);
    }

    // 3) 8 WMMAs on this wave's 32x64 sub-tile
#pragma unroll
    for (int mi = 0; mi < 2; ++mi)
#pragma unroll
      for (int ni = 0; ni < 4; ++ni)
        acc[mi][ni] = __builtin_amdgcn_wmma_f32_16x16x32_bf16(
            false, af[mi].v, false, bfb[ni].v, (short)0, acc[mi][ni], false, false);

    buf ^= 1;
  }

  // Epilogue. C/D layout: row = 8g + r, col = lane&15.
#pragma unroll
  for (int mi = 0; mi < 2; ++mi)
#pragma unroll
    for (int ni = 0; ni < 4; ++ni) {
      int ccol = col0 + wn + ni * 16 + l16;
      if (ccol >= N) continue;
      float bv = bias ? bias[ccol] : 0.0f;
#pragma unroll
      for (int r = 0; r < 8; ++r) {
        int crow = row0 + wm + mi * 16 + g * 8 + r;
        if (crow >= M) continue;
        float val = acc[mi][ni][r] * alpha + bv;
        if (Rb) val += Rb[(long long)crow * ldr + ccol];
        if (act == 1) {  // gelu (tanh approx, matches jax.nn.gelu default)
          float x = val;
          val = 0.5f * x * (1.0f + tanhf(0.7978845608f * (x + 0.044715f * x * x * x)));
        }
        Cb[(long long)crow * ldc + ccol] = val;
      }
    }
}

// ---------------- weight pre-pack (f32 -> bf16, once per launch) ----------------

__global__ void __launch_bounds__(256)
pack_bf16(const float* __restrict__ src, unsigned short* __restrict__ dst, long long n4)
{
  long long i = ((long long)blockIdx.x * 256 + threadIdx.x);
  if (i >= n4) return;
  float4 v = *(const float4*)(src + i * 4);
  uint2 q;
  q.x = ((unsigned int)f32_bf16(v.y) << 16) | f32_bf16(v.x);
  q.y = ((unsigned int)f32_bf16(v.w) << 16) | f32_bf16(v.z);
  *(uint2*)(dst + i * 4) = q;
}

// ---------------- wave-per-row kernels (wave32) ----------------

__global__ void __launch_bounds__(256)
softmax_rows(float* __restrict__ X, int cols, long long rows)
{
  long long row = (long long)blockIdx.x * 8 + (threadIdx.x >> 5);
  if (row >= rows) return;
  int lane = threadIdx.x & 31;
  float* R = X + row * (long long)cols;
  int cnt = cols >> 5;
  float v[32];
  float m = -1e30f;
  for (int i = 0; i < cnt; ++i) { v[i] = R[lane + i * 32]; m = fmaxf(m, v[i]); }
  for (int o = 16; o; o >>= 1) m = fmaxf(m, __shfl_xor(m, o));
  float s = 0.0f;
  for (int i = 0; i < cnt; ++i) { v[i] = __expf(v[i] - m); s += v[i]; }
  for (int o = 16; o; o >>= 1) s += __shfl_xor(s, o);
  float inv = 1.0f / s;
  for (int i = 0; i < cnt; ++i) R[lane + i * 32] = v[i] * inv;
}

__global__ void __launch_bounds__(256)
layernorm_rows(const float* __restrict__ X, const float* __restrict__ Y,
               const float* __restrict__ gam, const float* __restrict__ bet,
               float* __restrict__ O, int cols, long long rows)
{
  long long row = (long long)blockIdx.x * 8 + (threadIdx.x >> 5);
  if (row >= rows) return;
  int lane = threadIdx.x & 31;
  const float* xr = X + row * (long long)cols;
  const float* yr = Y ? Y + row * (long long)cols : nullptr;
  int cnt = cols >> 5;
  float v[32];
  float s = 0.0f;
  for (int i = 0; i < cnt; ++i) {
    int c = lane + i * 32;
    v[i] = xr[c] + (yr ? yr[c] : 0.0f);
    s += v[i];
  }
  for (int o = 16; o; o >>= 1) s += __shfl_xor(s, o);
  float mean = s / (float)cols;
  float var = 0.0f;
  for (int i = 0; i < cnt; ++i) { float d = v[i] - mean; var += d * d; }
  for (int o = 16; o; o >>= 1) var += __shfl_xor(var, o);
  var /= (float)cols;
  float rstd = rsqrtf(var + 1e-5f);
  for (int i = 0; i < cnt; ++i) {
    int c = lane + i * 32;
    O[row * (long long)cols + c] = (v[i] - mean) * rstd * gam[c] + bet[c];
  }
}

__global__ void __launch_bounds__(256)
gemv_sigmoid(const float* __restrict__ X, int ld, const float* __restrict__ wv,
             const float* __restrict__ bsc, float* __restrict__ out, int K, long long rows)
{
  long long row = (long long)blockIdx.x * 8 + (threadIdx.x >> 5);
  if (row >= rows) return;
  int lane = threadIdx.x & 31;
  const float* xr = X + row * (long long)ld;
  float s = 0.0f;
  for (int i = lane; i < K; i += 32) s += xr[i] * wv[i];
  for (int o = 16; o; o >>= 1) s += __shfl_xor(s, o);
  if (lane == 0) out[row] = 1.0f / (1.0f + __expf(-(s + bsc[0])));
}

// ---------------- elementwise kernels ----------------

__global__ void __launch_bounds__(256)
ew_iter_input(float* __restrict__ o, const float* __restrict__ base,
              const float* __restrict__ emb, const float* __restrict__ li,
              const float* __restrict__ hp, int D_, long long total)
{
  long long i = (long long)blockIdx.x * 256 + threadIdx.x;
  if (i >= total) return;
  float v = base[i] + emb[i % D_];
  if (li) v += li[i];
  if (hp) v += hp[i];
  o[i] = v;
}

__global__ void __launch_bounds__(256)
ew_gated_add(float* __restrict__ o, const float* __restrict__ h,
             const float* __restrict__ gate, const float* __restrict__ r,
             int D_, long long total)
{
  long long i = (long long)blockIdx.x * 256 + threadIdx.x;
  if (i >= total) return;
  o[i] = h[i] + gate[i / D_] * r[i];
}

__global__ void __launch_bounds__(256)
ew_gate_mul(float* __restrict__ o, const float* __restrict__ gate,
            const float* __restrict__ t, int D_, long long total)
{
  long long i = (long long)blockIdx.x * 256 + threadIdx.x;
  if (i >= total) return;
  o[i] = gate[i / D_] * t[i];
}

// ---------------------------------------------------------------------------
// Host orchestration
// ---------------------------------------------------------------------------

extern "C" void kernel_launch(void* const* d_in, const int* in_sizes, int n_in,
                              void* d_out, int out_size, void* d_ws, size_t ws_size,
                              hipStream_t stream) {
  (void)in_sizes; (void)n_in; (void)out_size; (void)ws_size;

  const int Bt = 2, S = 1024, D = 1024, DS = 320, Nslot = 512, H = 4, CH = 5, DFF = 4096;
  const int BS  = Bt * S;
  const int BN_ = Bt * Nslot;
  const int hd  = D / H;        // 256
  const int chd = DS / CH;      // 64
  const long long S2D = (long long)BS * D;
  const float rsD  = 0.03125f;
  const float rsHd = 0.0625f;
  const float rsDS = 0.05590169944f;
  const float rsCh = 0.125f;

  const float* x        = (const float*)d_in[0];
  const float* cache    = (const float*)d_in[1];
  const float* iter_emb = (const float*)d_in[2];
  const float* rq_w = (const float*)d_in[3];
  const float* rg_w = (const float*)d_in[4];
  const float* rg_b = (const float*)d_in[5];
  const float* ck_w = (const float*)d_in[6];
  const float* cv_w = (const float*)d_in[7];
  const float* aq_w = (const float*)d_in[8];
  const float* ak_w = (const float*)d_in[9];
  const float* av_w = (const float*)d_in[10];
  const float* ao_w = (const float*)d_in[11];
  const float* aq_b = (const float*)d_in[12];
  const float* ak_b = (const float*)d_in[13];
  const float* av_b = (const float*)d_in[14];
  const float* ao_b = (const float*)d_in[15];
  const float* f1_w = (const float*)d_in[16];
  const float* f1_b = (const float*)d_in[17];
  const float* f2_w = (const float*)d_in[18];
  const float* f2_b = (const float*)d_in[19];
  const float* n1_g = (const float*)d_in[20];
  const float* n1_b = (const float*)d_in[21];
  const float* n2_g = (const float*)d_in[22];
  const float* n2_b = (const float*)d_in[23];
  const float* sq_w = (const float*)d_in[24];
  const float* tk_w = (const float*)d_in[25];
  const float* tv_w = (const float*)d_in[26];
  const float* wg_w = (const float*)d_in[27];
  const float* wg_b = (const float*)d_in[28];
  const float* cq_w  = (const float*)d_in[29];
  const float* ck2_w = (const float*)d_in[30];
  const float* cv2_w = (const float*)d_in[31];
  const float* co_w  = (const float*)d_in[32];
  const float* cq_b  = (const float*)d_in[33];
  const float* ck2_b = (const float*)d_in[34];
  const float* cv2_b = (const float*)d_in[35];
  const float* co_b  = (const float*)d_in[36];
  const float* cn_g  = (const float*)d_in[37];
  const float* cn_b  = (const float*)d_in[38];
  // d_in[39] = max_iterations (== 2, compile-time constant here)

  size_t off = 0;
  auto allocB = [&](size_t bytes) -> void* {
    off = (off + 255) & ~(size_t)255;
    void* p = (char*)d_ws + off;
    off += bytes;
    return p;
  };
  auto allocF = [&](long long n) -> float* { return (float*)allocB((size_t)n * 4); };
  auto allocH = [&](long long n) -> unsigned short* { return (unsigned short*)allocB((size_t)n * 2); };

  float* hwork  = allocF(S2D);
  float* hc0    = allocF(S2D);
  float* henh   = allocF(S2D);
  float* hn     = allocF(S2D);
  float* qb     = allocF(S2D);
  float* kb     = allocF(S2D);
  float* vb     = allocF(S2D);
  float* attn_o = allocF(S2D);
  float* attnb  = allocF(S2D);
  float* big    = allocF((long long)BS * DFF);
  float* gate   = allocF(BS);
  float* wgate  = allocF(BS);
  float* ccA    = allocF((long long)BN_ * DS);
  float* ccB    = allocF((long long)BN_ * DS);
  float* sqb    = allocF((long long)BN_ * DS);
  float* tkb    = allocF((long long)BS * DS);
  float* tvb    = allocF((long long)BS * DS);
  float* gtvb   = allocF((long long)BS * DS);
  float* cqb    = allocF((long long)BN_ * DS);
  float* ckb    = allocF((long long)BN_ * DS);
  float* cvb    = allocF((long long)BN_ * DS);
  float* cob    = allocF((long long)BN_ * DS);
  float* cattnb = allocF((long long)BN_ * DS);

  // bf16 weight copies
  unsigned short* w_rq = allocH((long long)D * D);
  unsigned short* w_ck = allocH((long long)DS * D);
  unsigned short* w_cv = allocH((long long)DS * D);
  unsigned short* w_aq = allocH((long long)D * D);
  unsigned short* w_ak = allocH((long long)D * D);
  unsigned short* w_av = allocH((long long)D * D);
  unsigned short* w_ao = allocH((long long)D * D);
  unsigned short* w_f1 = allocH((long long)D * DFF);
  unsigned short* w_f2 = allocH((long long)DFF * D);
  unsigned short* w_sq = allocH((long long)DS * DS);
  unsigned short* w_tk = allocH((long long)D * DS);
  unsigned short* w_tv = allocH((long long)D * DS);
  unsigned short* w_cq = allocH((long long)DS * DS);
  unsigned short* w_k2 = allocH((long long)DS * DS);
  unsigned short* w_v2 = allocH((long long)DS * DS);
  unsigned short* w_co = allocH((long long)DS * DS);

  auto pack = [&](const float* src, unsigned short* dst, long long n) {
    long long n4 = n >> 2;
    pack_bf16<<<dim3((unsigned)((n4 + 255) / 256)), dim3(256), 0, stream>>>(src, dst, n4);
  };
  pack(rq_w, w_rq, (long long)D * D);   pack(ck_w, w_ck, (long long)DS * D);
  pack(cv_w, w_cv, (long long)DS * D);  pack(aq_w, w_aq, (long long)D * D);
  pack(ak_w, w_ak, (long long)D * D);   pack(av_w, w_av, (long long)D * D);
  pack(ao_w, w_ao, (long long)D * D);   pack(f1_w, w_f1, (long long)D * DFF);
  pack(f2_w, w_f2, (long long)DFF * D); pack(sq_w, w_sq, (long long)DS * DS);
  pack(tk_w, w_tk, (long long)D * DS);  pack(tv_w, w_tv, (long long)D * DS);
  pack(cq_w, w_cq, (long long)DS * DS); pack(ck2_w, w_k2, (long long)DS * DS);
  pack(cv2_w, w_v2, (long long)DS * DS); pack(co_w, w_co, (long long)DS * DS);

  auto gemm = [&](const float* A, int lda, long long sA, long long hA,
                  const void* Bm, int ldb, long long sB, long long hB, int transB, int bBf16,
                  const float* bias,
                  const float* R, int ldr, long long sR, long long hR,
                  float* C, int ldc, long long sC, long long hC,
                  int M, int N, int K, float alpha, int act, int nZ, int nH) {
    dim3 grid((N + TILE_N - 1) / TILE_N, (M + TILE_M - 1) / TILE_M, nZ);
    gemm_bf16_wmma<<<grid, dim3(256), 0, stream>>>(
        A, lda, sA, hA, Bm, ldb, sB, hB, transB, bBf16, bias,
        R, ldr, sR, hR, C, ldc, sC, hC, M, N, K, alpha, act, nH);
  };
  auto softmaxL = [&](float* X, int cols, long long rows) {
    softmax_rows<<<dim3((unsigned)((rows + 7) / 8)), dim3(256), 0, stream>>>(X, cols, rows);
  };
  auto lnL = [&](const float* X, const float* Y, const float* gm, const float* bt,
                 float* O, int cols, long long rows) {
    layernorm_rows<<<dim3((unsigned)((rows + 7) / 8)), dim3(256), 0, stream>>>(X, Y, gm, bt, O, cols, rows);
  };

  const long long Z = 0;
  const float* ccSrc = cache;

  for (int it = 0; it < 2; ++it) {
    const bool last = (it == 1);

    ew_iter_input<<<dim3((unsigned)((S2D + 255) / 256)), dim3(256), 0, stream>>>(
        hwork, it == 0 ? x : hc0, iter_emb + (long long)it * D,
        it == 0 ? nullptr : x, it == 0 ? nullptr : hc0, D, S2D);

    // ---------------- mem_read ----------------
    gemm(hwork, D, Z, Z, w_rq, D, Z, Z, 0, 1, nullptr, nullptr, 0, Z, Z,
         qb, D, Z, Z, BS, D, D, 1.f, 0, 1, 1);                               // q = h @ rq_w
    gemm(ccSrc, DS, Z, Z, w_ck, D, Z, Z, 0, 1, nullptr, nullptr, 0, Z, Z,
         kb, D, Z, Z, BN_, D, DS, 1.f, 0, 1, 1);                             // k = cc @ ck_w
    gemm(ccSrc, DS, Z, Z, w_cv, D, Z, Z, 0, 1, nullptr, nullptr, 0, Z, Z,
         vb, D, Z, Z, BN_, D, DS, 1.f, 0, 1, 1);                             // v = cc @ cv_w
    gemm(qb, D, (long long)S * D, Z, kb, D, (long long)Nslot * D, Z, 1, 0, nullptr,
         nullptr, 0, Z, Z, big, Nslot, (long long)S * Nslot, Z,
         S, Nslot, D, rsD, 0, Bt, 1);                                        // q k^T
    softmaxL(big, Nslot, (long long)Bt * S);
    gemm(big, Nslot, (long long)S * Nslot, Z, vb, D, (long long)Nslot * D, Z, 0, 0, nullptr,
         nullptr, 0, Z, Z, attnb, D, (long long)S * D, Z,
         S, D, Nslot, 1.f, 0, Bt, 1);                                        // readout
    gemv_sigmoid<<<dim3((BS + 7) / 8), dim3(256), 0, stream>>>(hwork, D, rg_w, rg_b, gate, D, BS);
    ew_gated_add<<<dim3((unsigned)((S2D + 255) / 256)), dim3(256), 0, stream>>>(
        henh, hwork, gate, attnb, D, S2D);                                   // h_enh

    // ---------------- compute_block ----------------
    lnL(henh, nullptr, n1_g, n1_b, hn, D, BS);
    gemm(hn, D, Z, Z, w_aq, D, Z, Z, 0, 1, aq_b, nullptr, 0, Z, Z,
         qb, D, Z, Z, BS, D, D, 1.f, 0, 1, 1);
    gemm(hn, D, Z, Z, w_ak, D, Z, Z, 0, 1, ak_b, nullptr, 0, Z, Z,
         kb, D, Z, Z, BS, D, D, 1.f, 0, 1, 1);
    gemm(hn, D, Z, Z, w_av, D, Z, Z, 0, 1, av_b, nullptr, 0, Z, Z,
         vb, D, Z, Z, BS, D, D, 1.f, 0, 1, 1);
    gemm(qb, D, (long long)S * D, (long long)hd, kb, D, (long long)S * D, (long long)hd, 1, 0,
         nullptr, nullptr, 0, Z, Z,
         big, S, (long long)H * S * S, (long long)S * S,
         S, S, hd, rsHd, 0, Bt * H, H);                                      // QK^T per (b,h)
    softmaxL(big, S, (long long)Bt * H * S);
    gemm(big, S, (long long)H * S * S, (long long)S * S, vb, D, (long long)S * D, (long long)hd, 0, 0,
         nullptr, nullptr, 0, Z, Z,
         attn_o, D, (long long)S * D, (long long)hd,
         S, hd, S, 1.f, 0, Bt * H, H);                                       // P @ V
    gemm(attn_o, D, Z, Z, w_ao, D, Z, Z, 0, 1, ao_b, nullptr, 0, Z, Z,
         attnb, D, Z, Z, BS, D, D, 1.f, 0, 1, 1);
    lnL(henh, attnb, n1_g, n1_b, hwork, D, BS);                              // h = LN(h_enh+attn)
    gemm(hwork, D, Z, Z, w_f1, DFF, Z, Z, 0, 1, f1_b, nullptr, 0, Z, Z,
         big, DFF, Z, Z, BS, DFF, D, 1.f, 1, 1, 1);                          // gelu ffn1
    gemm(big, DFF, Z, Z, w_f2, D, Z, Z, 0, 1, f2_b, nullptr, 0, Z, Z,
         attn_o, D, Z, Z, BS, D, DFF, 1.f, 0, 1, 1);                         // ffn2
    float* hout = last ? (float*)d_out : hc0;
    lnL(hwork, attn_o, n2_g, n2_b, hout, D, BS);                             // h_comp

    if (!last) {
      // ---------------- mem_write ----------------
      gemm(ccSrc, DS, Z, Z, w_sq, DS, Z, Z, 0, 1, nullptr, nullptr, 0, Z, Z,
           sqb, DS, Z, Z, BN_, DS, DS, 1.f, 0, 1, 1);
      gemm(hout, D, Z, Z, w_tk, DS, Z, Z, 0, 1, nullptr, nullptr, 0, Z, Z,
           tkb, DS, Z, Z, BS, DS, D, 1.f, 0, 1, 1);
      gemm(hout, D, Z, Z, w_tv, DS, Z, Z, 0, 1, nullptr, nullptr, 0, Z, Z,
           tvb, DS, Z, Z, BS, DS, D, 1.f, 0, 1, 1);
      gemv_sigmoid<<<dim3((BS + 7) / 8), dim3(256), 0, stream>>>(hout, D, wg_w, wg_b, wgate, D, BS);
      ew_gate_mul<<<dim3((unsigned)(((long long)BS * DS + 255) / 256)), dim3(256), 0, stream>>>(
          gtvb, wgate, tvb, DS, (long long)BS * DS);
      gemm(sqb, DS, (long long)Nslot * DS, Z, tkb, DS, (long long)S * DS, Z, 1, 0, nullptr,
           nullptr, 0, Z, Z, big, S, (long long)Nslot * S, Z,
           Nslot, S, DS, rsDS, 0, Bt, 1);
      softmaxL(big, S, (long long)Bt * Nslot);
      gemm(big, S, (long long)Nslot * S, Z, gtvb, DS, (long long)S * DS, Z, 0, 0, nullptr,
           ccSrc, DS, (long long)Nslot * DS, Z,
           ccA, DS, (long long)Nslot * DS, Z,
           Nslot, DS, S, 1.f, 0, Bt, 1);                                     // cc + upd

      // ---------------- cache_self_attn ----------------
      gemm(ccA, DS, Z, Z, w_cq, DS, Z, Z, 0, 1, cq_b, nullptr, 0, Z, Z,
           cqb, DS, Z, Z, BN_, DS, DS, 1.f, 0, 1, 1);
      gemm(ccA, DS, Z, Z, w_k2, DS, Z, Z, 0, 1, ck2_b, nullptr, 0, Z, Z,
           ckb, DS, Z, Z, BN_, DS, DS, 1.f, 0, 1, 1);
      gemm(ccA, DS, Z, Z, w_v2, DS, Z, Z, 0, 1, cv2_b, nullptr, 0, Z, Z,
           cvb, DS, Z, Z, BN_, DS, DS, 1.f, 0, 1, 1);
      gemm(cqb, DS, (long long)Nslot * DS, (long long)chd,
           ckb, DS, (long long)Nslot * DS, (long long)chd, 1, 0, nullptr,
           nullptr, 0, Z, Z,
           big, Nslot, (long long)CH * Nslot * Nslot, (long long)Nslot * Nslot,
           Nslot, Nslot, chd, rsCh, 0, Bt * CH, CH);
      softmaxL(big, Nslot, (long long)Bt * CH * Nslot);
      gemm(big, Nslot, (long long)CH * Nslot * Nslot, (long long)Nslot * Nslot,
           cvb, DS, (long long)Nslot * DS, (long long)chd, 0, 0,
           nullptr, nullptr, 0, Z, Z,
           cob, DS, (long long)Nslot * DS, (long long)chd,
           Nslot, chd, Nslot, 1.f, 0, Bt * CH, CH);
      gemm(cob, DS, Z, Z, w_co, DS, Z, Z, 0, 1, co_b, nullptr, 0, Z, Z,
           cattnb, DS, Z, Z, BN_, DS, DS, 1.f, 0, 1, 1);
      lnL(ccA, cattnb, cn_g, cn_b, ccB, DS, BN_);
      ccSrc = ccB;
    }
  }
}
